// EER_GCN_9199819948207
// MI455X (gfx1250) — matrix operand order repaired and checked by reading, not compiled, not run on hardware
//
#include <hip/hip_runtime.h>
#include <hip/hip_bf16.h>
#include <math.h>

typedef __attribute__((ext_vector_type(16))) _Float16 v16h;
typedef __attribute__((ext_vector_type(8)))  float    v8f;
typedef __attribute__((ext_vector_type(2)))  float    v2f;
typedef __attribute__((ext_vector_type(4)))  unsigned int u32x4;
typedef __attribute__((ext_vector_type(8)))  int      i32x8;
typedef __attribute__((ext_vector_type(4)))  int      i32x4;

#define N_FEAT  5
#define BATCH   4096
#define SLOPE   0.15f
#define BN_EPS  1e-5f
#define KTILES  62          // 1984 / 32
#define BPB     16          // batch elements per block
#define XCH     5120        // one padded X chunk: 64 rows x 80 cols

// workspace layout (float offsets)
#define OFF_BNSUM    0
#define OFF_BNSQ     8
#define OFF_BNSCALE 16
#define OFF_BNSHIFT 24
#define OFF_LHAT    64              // padded 64x64 L_hat (4096 floats)
#define OFF_WB_BYTES (8192*4)       // f16 WMMA-layout fc1 weights start here

#if __has_builtin(__builtin_amdgcn_wmma_f32_16x16x4_f32)
#define HAVE_WMMA_F32X4 1
#else
#define HAVE_WMMA_F32X4 0
#endif
#if __has_builtin(__builtin_amdgcn_tensor_load_to_lds) && \
    __has_builtin(__builtin_amdgcn_s_wait_tensorcnt)
#define HAVE_TDM 1
#else
#define HAVE_TDM 0
#endif

__device__ __forceinline__ float leaky(float v) { return v >= 0.f ? v : SLOPE * v; }

// ---------------- K0: zero accumulators + repack fc1_W (f32 [1984][128]) into
// f16 WMMA B-fragment layout: frag[(ntile*62+kt)*32 + lane][16 halves] --------
__global__ void k0_prep(const float* __restrict__ fc1W, float* __restrict__ ws,
                        float* __restrict__ out0) {
    _Float16* wB = (_Float16*)((char*)ws + OFF_WB_BYTES);
    int idx = blockIdx.x * blockDim.x + threadIdx.x;
    int stride = gridDim.x * blockDim.x;
    if (idx < 16) ws[OFF_BNSUM + idx] = 0.f;
    if (idx == 0) out0[0] = 0.f;
    for (int i = idx; i < 1984 * 128; i += stride) {
        int r = i >> 7, c = i & 127;
        int ntile = c >> 4, n = c & 15;
        int kt = r >> 5,  k = r & 31;
        int lane = n + (((k >> 3) & 1) << 4);    // B16: lane = n + 16*((k/8)%2)
        int half = ((k >> 4) << 3) | (k & 7);    // B16: half = 8*(k/16) + k%8
        wB[(((ntile * KTILES + kt) * 32 + lane) << 4) + half] = (_Float16)fc1W[i];
    }
}

// ---------------- K1: BatchNorm batch stats ---------------------------------
__global__ void k1_bnstats(const float* __restrict__ x, int rows,
                           float* __restrict__ ws) {
    __shared__ float sAcc[10];
    if (threadIdx.x < 10) sAcc[threadIdx.x] = 0.f;
    __syncthreads();
    float s[5] = {0,0,0,0,0}, q[5] = {0,0,0,0,0};
    int idx = blockIdx.x * blockDim.x + threadIdx.x;
    int stride = gridDim.x * blockDim.x;
    for (int r = idx; r < rows; r += stride) {
        const float* p = x + (size_t)r * 5;
#pragma unroll
        for (int f = 0; f < 5; ++f) { float v = p[f]; s[f] += v; q[f] += v * v; }
    }
#pragma unroll
    for (int f = 0; f < 5; ++f) {
        atomicAdd(&sAcc[f],     s[f]);
        atomicAdd(&sAcc[5 + f], q[f]);
    }
    __syncthreads();
    if (threadIdx.x < 5)       atomicAdd(&ws[OFF_BNSUM + threadIdx.x], sAcc[threadIdx.x]);
    else if (threadIdx.x < 10) atomicAdd(&ws[OFF_BNSQ + threadIdx.x - 5], sAcc[threadIdx.x]);
}

// ---------------- K2 (1 block): BN finalize, lambda_max (shifted power
// iteration on dense L = D - A), padded 64x64 L_hat -> workspace -------------
__global__ void k2_graph(const int* __restrict__ eidx, const float* __restrict__ ew,
                         int E, const float* __restrict__ gamma,
                         const float* __restrict__ beta, int rows,
                         float* __restrict__ ws) {
    __shared__ float sM[62 * 62];
    __shared__ float sv[64], svn[64], sred[64], sdeg[64];
    __shared__ float sShift;
    int tid = threadIdx.x;                       // blockDim = 64

    if (tid < 5) {
        float cnt = (float)rows;
        float mu  = ws[OFF_BNSUM + tid] / cnt;
        float var = ws[OFF_BNSQ + tid] / cnt - mu * mu;
        float sc  = gamma[tid] * rsqrtf(var + BN_EPS);
        ws[OFF_BNSCALE + tid] = sc;
        ws[OFF_BNSHIFT + tid] = beta[tid] - mu * sc;
    }
    for (int i = tid; i < 3844; i += 64) sM[i] = 0.f;
    __syncthreads();
    for (int e = tid; e < E; e += 64) {
        int s = eidx[e], d = eidx[E + e];
        sM[s * 62 + d] = ew[e];
    }
    __syncthreads();
    float lo = 1e30f;
    if (tid < 62) {
        float dsum = 0.f;
        for (int j = 0; j < 62; ++j) dsum += sM[tid * 62 + j];
        for (int j = 0; j < 62; ++j) sM[tid * 62 + j] = -sM[tid * 62 + j];
        sM[tid * 62 + tid] += dsum;
        float rad = 0.f;
        for (int j = 0; j < 62; ++j) if (j != tid) rad += fabsf(sM[tid * 62 + j]);
        lo = sM[tid * 62 + tid] - rad;
    }
    sred[tid] = lo;
    __syncthreads();
    for (int off = 32; off > 0; off >>= 1) {
        if (tid < off) sred[tid] = fminf(sred[tid], sred[tid + off]);
        __syncthreads();
    }
    if (tid == 0) sShift = fmaxf(0.f, -sred[0]);
    __syncthreads();
    float shift = sShift;
    sv[tid] = (tid < 62) ? 0.126745f : 0.f;
    __syncthreads();
    float lastn = 1.f;
    for (int it = 0; it < 64; ++it) {
        float acc = 0.f;
        if (tid < 62) {
            acc = shift * sv[tid];
            for (int j = 0; j < 62; ++j) acc += sM[tid * 62 + j] * sv[j];
        }
        svn[tid] = acc;
        sred[tid] = acc * acc;
        __syncthreads();
        for (int off = 32; off > 0; off >>= 1) {
            if (tid < off) sred[tid] += sred[tid + off];
            __syncthreads();
        }
        float nrm = sqrtf(sred[0]);
        lastn = nrm;
        float inv = (nrm > 0.f) ? 1.f / nrm : 0.f;
        sv[tid] = svn[tid] * inv;
        __syncthreads();
    }
    float lam = lastn - shift;
    if (!(lam > 1e-3f)) lam = 2.f;
    float scale = 2.f / lam;

    __syncthreads();
    for (int i = tid; i < 3844; i += 64) sM[i] = 0.f;
    __syncthreads();
    for (int e = tid; e < E; e += 64) {
        int s = eidx[e], d = eidx[E + e];
        sM[s * 62 + d] = leaky(ew[e]);
    }
    __syncthreads();
    if (tid < 62) {
        float dsum = 0.f;
        for (int j = 0; j < 62; ++j) dsum += sM[tid * 62 + j];
        sdeg[tid] = dsum;
    }
    __syncthreads();
    // padded 64x64 L_hat: zeros in rows/cols 62,63
    for (int j = 0; j < 64; ++j) {
        float v = 0.f;
        if (tid < 62 && j < 62)
            v = (j == tid) ? (sdeg[tid] * scale - 1.f) : (-sM[j * 62 + tid] * scale);
        ws[OFF_LHAT + tid * 64 + j] = v;
    }
}

#if HAVE_TDM
// One TDM descriptor: copy 4096 contiguous f32 (padded L_hat) global -> LDS.
__device__ __forceinline__ void tdm_load_lhat(const float* gsrc, void* lds_dst) {
    unsigned lds = (unsigned)(size_t)lds_dst;
    unsigned long long ga = (unsigned long long)(size_t)gsrc;
    u32x4 g0;
    g0[0] = 1u;                                   // count=1 (valid user D#)
    g0[1] = lds;                                  // lds_addr
    g0[2] = (unsigned)(ga & 0xffffffffu);         // global_addr[31:0]
    g0[3] = (unsigned)((ga >> 32) & 0x01ffffffu) | (2u << 30);  // addr[56:32]|type=2
    i32x8 g1;
    g1[0] = (int)(2u << 16);                      // data_size=4B, mask/flags=0
    g1[1] = (int)(0x1000u << 16);                 // tensor_dim0=4096 (bits 63:48)
    g1[2] = (int)(1u << 16);                      // tensor_dim0 hi=0 | tensor_dim1=1
    g1[3] = (int)(0x1000u << 16);                 // tensor_dim1 hi=0 | tile_dim0=4096
    g1[4] = 1;                                    // tile_dim1=1, tile_dim2=0
    g1[5] = 4096;                                 // tensor_dim0_stride lo
    g1[6] = 0;
    g1[7] = 0;
    i32x4 gz = {0, 0, 0, 0};
#if defined(__clang_major__) && __clang_major__ >= 23
    i32x8 gz8 = {0, 0, 0, 0, 0, 0, 0, 0};
    __builtin_amdgcn_tensor_load_to_lds(g0, g1, gz, gz, gz8, 0);
#else
    __builtin_amdgcn_tensor_load_to_lds(g0, g1, gz, gz, 0);
#endif
}
#endif

// ---------------- K3: fully fused per-16-batch pipeline ----------------------
__global__ __launch_bounds__(256, 1)
void k3_main(const float* __restrict__ x, const int* __restrict__ y,
             const float* __restrict__ chebW, const float* __restrict__ chebB,
             const float* __restrict__ fc1b, const float* __restrict__ fc2W,
             const float* __restrict__ fc2b, const float* __restrict__ ws,
             float* __restrict__ out) {
    __shared__ float sLhat[64 * 64];              // padded
    __shared__ float sXall[3 * XCH];              // T0|T1|T2 as [64 x 80] each
    __shared__ __align__(16) _Float16 sAfrag[KTILES * 32 * 16];
    __shared__ float sY[BPB * 128];
    __shared__ float sCWp[16 * 32];               // chebW padded to 16 K-rows
    __shared__ float sCB[32], sF1B[128], sBNs[5], sBNb[5], sLoss[BPB];
    int tid = threadIdx.x;
    int b0  = blockIdx.x * BPB;
    float* sX0 = sXall;
    float* sX1 = sXall + XCH;
    float* sX2 = sXall + 2 * XCH;

#if HAVE_TDM
    if (tid < 32) {                               // wave 0 drives the TDM DMA
        tdm_load_lhat(ws + OFF_LHAT, (void*)&sLhat[0]);
        __builtin_amdgcn_s_wait_tensorcnt(0);
    }
#else
    for (int i = tid; i < 4096; i += 256) sLhat[i] = ws[OFF_LHAT + i];
#endif
    for (int i = tid; i < 480; i += 256) sCWp[i] = chebW[i];
    if (tid < 32)  sCWp[480 + tid] = 0.f;         // zero K-row 15
    if (tid < 32)  sCB[tid]  = chebB[tid];
    if (tid < 128) sF1B[tid] = fc1b[tid];
    if (tid < 5) { sBNs[tid] = ws[OFF_BNSCALE + tid]; sBNb[tid] = ws[OFF_BNSHIFT + tid]; }
    if (tid < 160) sX0[4960 + tid] = 0.f;         // zero pad rows 62,63 of X0
    __syncthreads();

    // stage 1: BN + leaky -> X0[i][b*5+f]  (coalesced in global idx)
    for (int idx = tid; idx < BPB * 310; idx += 256) {
        int b = idx / 310, r = idx % 310, i = r / 5, f = r % 5;
        float v = x[(size_t)(b0 + b) * 310 + r];
        sX0[i * 80 + b * 5 + f] = leaky(v * sBNs[f] + sBNb[f]);
    }
    __syncthreads();

#if HAVE_WMMA_F32X4
    // stages 2/3: X1 = Lhat@X0, X2 = 2*Lhat@X1 - X0 via WMMA f32 16x16x4.
    // All operands padded -> no guards; pad rows self-zero through the GEMM.
    {
        int wv = tid >> 5, lane = tid & 31;
        int lr = lane & 15, g = lane >> 4;
        for (int pass = 0; pass < 2; ++pass) {
            const float* src = pass ? sX1 : sX0;
            float*       dst = pass ? sX2 : sX1;
            for (int t = wv; t < 20; t += 8) {    // wave-uniform tile loop
                int mt = t / 5, nt = t % 5;
                int row = mt * 16 + lr;           // 0..63 (padded)
                int col = nt * 16 + lr;           // 0..79
                v8f acc = {};
#pragma unroll
                for (int kk = 0; kk < 16; ++kk) {
                    int ka = kk * 4 + 2 * g;
                    v2f a, b;
                    a[0] = sLhat[(row << 6) + ka];
                    a[1] = sLhat[(row << 6) + ka + 1];
                    b[0] = src[ka * 80 + col];
                    b[1] = src[(ka + 1) * 80 + col];
                    acc = __builtin_amdgcn_wmma_f32_16x16x4_f32(
                              false, a, false, b, (short)0, acc, false, false);
                }
#pragma unroll
                for (int r = 0; r < 8; ++r) {
                    int m = mt * 16 + r + 8 * g;  // 0..63, pad rows get zeros
                    int o = m * 80 + col;
                    dst[o] = pass ? (2.f * acc[r] - sX0[o]) : acc[r];
                }
            }
            __syncthreads();
        }
    }
#else
    for (int idx = tid; idx < 4960; idx += 256) {
        int i = idx / 80, col = idx % 80;
        float acc = 0.f;
        for (int j = 0; j < 62; ++j) acc += sLhat[(i << 6) + j] * sX0[j * 80 + col];
        sX1[idx] = acc;
    }
    if (tid < 160) sX1[4960 + tid] = 0.f;
    __syncthreads();
    for (int idx = tid; idx < 4960; idx += 256) {
        int i = idx / 80, col = idx % 80;
        float acc = 0.f;
        for (int j = 0; j < 62; ++j) acc += sLhat[(i << 6) + j] * sX1[j * 80 + col];
        sX2[idx] = 2.f * acc - sX0[idx];
    }
    if (tid < 160) sX2[4960 + tid] = 0.f;
    __syncthreads();
#endif

#if HAVE_WMMA_F32X4
    // stage 4: Cheb einsum as [992 x 16] @ [16 x 32] WMMA f32 GEMM; K row 15 is
    // zero in B, A's k=15 element uses clamped-address load + select.
    {
        int wv = tid >> 5, lane = tid & 31;
        int lr = lane & 15, g = lane >> 4;
        for (int t = wv; t < 124; t += 8) {       // 62 M-tiles x 2 N-tiles
            int mt = t >> 1, nt = t & 1;
            int arow = mt * 16 + lr;              // (b*62+i) row, < 992
            int ab = arow / 62, ai = arow - ab * 62;
            int abase = ai * 80 + ab * 5;         // within one X chunk
            int ncol = nt * 16 + lr;              // hid column, < 32
            v8f acc = {};
#pragma unroll
            for (int kk = 0; kk < 4; ++kk) {
                int ka = kk * 4 + 2 * g;
                int c0 = ka / 5,       f0 = ka - c0 * 5;
                int k1 = ka + 1;
                int c1 = k1 / 5,       f1 = k1 - c1 * 5;
                int c1c = (c1 > 2) ? 2 : c1;      // clamp address, select value
                v2f a, b;
                a[0] = sXall[c0 * XCH + abase + f0];
                float a1 = sXall[c1c * XCH + abase + f1];
                a[1] = (k1 < 15) ? a1 : 0.f;
                b[0] = sCWp[ka * 32 + ncol];
                b[1] = sCWp[k1 * 32 + ncol];      // row 15 is zeroed
                acc = __builtin_amdgcn_wmma_f32_16x16x4_f32(
                          false, a, false, b, (short)0, acc, false, false);
            }
#pragma unroll
            for (int r = 0; r < 8; ++r) {
                int m = mt * 16 + r + 8 * g;      // < 992
                int b = m / 62, i = m - b * 62;
                float v = leaky(acc[r] + sCB[ncol]);
                int k = ncol;
                int lane2 = b + (((k >> 3) & 1) << 4);   // f16 A-frag mapping
                int half  = ((k >> 4) << 3) | (k & 7);
                sAfrag[((i * 32 + lane2) << 4) + half] = (_Float16)v;
            }
        }
    }
#else
    for (int idx = tid; idx < BPB * 62 * 32; idx += 256) {
        int b = idx / (62 * 32), r = idx % (62 * 32), i = r >> 5, hh = r & 31;
        int base = i * 80 + b * 5;
        float acc = sCB[hh];
#pragma unroll
        for (int f = 0; f < 5; ++f)
            acc += sX0[base + f] * sCWp[f * 32 + hh]
                 + sX1[base + f] * sCWp[(5 + f) * 32 + hh]
                 + sX2[base + f] * sCWp[(10 + f) * 32 + hh];
        acc = leaky(acc);
        int k = hh;
        int lane = b + (((k >> 3) & 1) << 4);
        int half = ((k >> 4) << 3) | (k & 7);
        sAfrag[((i * 32 + lane) << 4) + half] = (_Float16)acc;
    }
#endif
    __syncthreads();

    // stage 5: fc1 [16 x 1984] @ [1984 x 128] via WMMA f16, 8 waves x 16 cols
    {
        int wv = tid >> 5, lane = tid & 31;
        const _Float16* wB = (const _Float16*)((const char*)ws + OFF_WB_BYTES);
        const _Float16* bptr = wB + ((size_t)(wv * KTILES) * 32 + lane) * 16;
        v8f acc = {};
#pragma unroll 2
        for (int kt = 0; kt < KTILES; ++kt) {
            v16h a  = *(const v16h*)(&sAfrag[(kt * 32 + lane) << 4]);
            v16h bf = *(const v16h*)(bptr + (size_t)kt * 32 * 16);
            __builtin_prefetch(bptr + (size_t)(kt + 1) * 32 * 16, 0, 1);
            acc = __builtin_amdgcn_wmma_f32_16x16x32_f16(
                      false, a, false, bf, (short)0, acc, false, false);
        }
#pragma unroll
        for (int r = 0; r < 8; ++r) {
            int m = r + ((lane >> 4) << 3);
            int n = (wv << 4) + (lane & 15);
            sY[m * 128 + n] = leaky(acc[r] + sF1B[n]);
        }
    }
    __syncthreads();

    // stage 6: fc2 + log_softmax + logits + loss
    if (tid < BPB) {
        int b = tid;
        float p0 = fc2b[0], p1 = fc2b[1], p2 = fc2b[2];
        const float* yr = &sY[b * 128];
        for (int j = 0; j < 128; ++j) {
            float v = yr[j];
            p0 += v * fc2W[j * 3 + 0];
            p1 += v * fc2W[j * 3 + 1];
            p2 += v * fc2W[j * 3 + 2];
        }
        float m  = fmaxf(p0, fmaxf(p1, p2));
        float e0 = expf(p0 - m), e1 = expf(p1 - m), e2 = expf(p2 - m);
        float lse = m + logf(e0 + e1 + e2);
        float lp0 = p0 - lse, lp1 = p1 - lse, lp2 = p2 - lse;
        int bg = b0 + b;
        out[1 + bg * 3 + 0] = expf(lp0);
        out[1 + bg * 3 + 1] = expf(lp1);
        out[1 + bg * 3 + 2] = expf(lp2);
        int yy = y[bg];
        sLoss[b] = -((yy == 0) ? lp0 : (yy == 1) ? lp1 : lp2);
    }
    __syncthreads();
    if (tid == 0) {
        float t = 0.f;
        for (int b = 0; b < BPB; ++b) t += sLoss[b];
        atomicAdd(out, t * (1.0f / BATCH));
    }
}

extern "C" void kernel_launch(void* const* d_in, const int* in_sizes, int n_in,
                              void* d_out, int out_size, void* d_ws, size_t ws_size,
                              hipStream_t stream) {
    const float* x    = (const float*)d_in[0];
    const int*   yv   = (const int*)  d_in[1];
    const int*   eidx = (const int*)  d_in[2];
    const float* ew   = (const float*)d_in[3];
    const float* gam  = (const float*)d_in[4];
    const float* bet  = (const float*)d_in[5];
    const float* cW   = (const float*)d_in[6];
    const float* cB   = (const float*)d_in[7];
    const float* f1W  = (const float*)d_in[8];
    const float* f1b  = (const float*)d_in[9];
    const float* f2W  = (const float*)d_in[10];
    const float* f2b  = (const float*)d_in[11];
    float* out = (float*)d_out;
    float* ws  = (float*)d_ws;
    int E    = in_sizes[3];
    int rows = in_sizes[0] / N_FEAT;

    hipLaunchKernelGGL(k0_prep,    dim3(256), dim3(256), 0, stream, f1W, ws, out);
    hipLaunchKernelGGL(k1_bnstats, dim3(256), dim3(256), 0, stream, x, rows, ws);
    hipLaunchKernelGGL(k2_graph,   dim3(1),   dim3(64),  0, stream, eidx, ew, E, gam, bet, rows, ws);
    hipLaunchKernelGGL(k3_main,    dim3(BATCH / BPB), dim3(256), 0, stream,
                       x, yv, cW, cB, f1b, f2W, f2b, ws, out);
}
